// ComposedStateMixing_18184891531316
// MI455X (gfx1250) — compile-verified, matrix operand order checked by reading, not run on hardware
//
#include <hip/hip_runtime.h>
#include <hip/hip_bf16.h>

#define T_SEQ 2048
#define DMODEL 512
#define NH 8
#define HDIM 64
#define TD (T_SEQ * DMODEL)
#define EPSV 1e-8f
#define LDT 72                 // padded LDS row stride (elements) -> conflict-free frag rows
#define TILE_E (64 * LDT)
#define NCHUNK (T_SEQ / 64)

typedef __bf16 bf16_t;
typedef __attribute__((ext_vector_type(16))) __bf16 v16bf;
typedef __attribute__((ext_vector_type(8)))  __bf16 v8bf;
typedef __attribute__((ext_vector_type(8)))  float  v8f;
typedef __attribute__((ext_vector_type(4)))  unsigned v4u;
typedef __attribute__((ext_vector_type(8)))  unsigned v8u;

union FragU {
    v16bf v;
    v8bf  h[2];
    bf16_t e[16];
    unsigned short s[16];
};

__device__ inline v8f zero8() {
    v8f z = {0.f, 0.f, 0.f, 0.f, 0.f, 0.f, 0.f, 0.f};
    return z;
}

__device__ inline v8f wmma_bf16(v16bf a, v16bf b, v8f c) {
    return __builtin_amdgcn_wmma_f32_16x16x32_bf16(
        false, a, false, b, (short)0, c, false, false);
}

// 16x32 fragment: lane m=lane&15 supplies row m; K runs at {half*8..+7} and
// {16+half*8..+7} relative to base.
__device__ inline FragU load_fragu(const bf16_t* base, int ld, int lane) {
    int m = lane & 15, half = (lane >> 4) & 1;
    const bf16_t* p = base + (long)m * ld + half * 8;
    FragU u;
    u.h[0] = *(const v8bf*)(p);
    u.h[1] = *(const v8bf*)(p + 16);
    return u;
}

__device__ inline FragU load_fragu_f32(const float* base, int ld, int lane) {
    int m = lane & 15, half = (lane >> 4) & 1;
    const float* p = base + (long)m * ld + half * 8;
    FragU u;
#pragma unroll
    for (int j = 0; j < 8; ++j) u.e[j] = (bf16_t)p[j];
#pragma unroll
    for (int j = 0; j < 8; ++j) u.e[8 + j] = (bf16_t)p[16 + j];
    return u;
}

__device__ inline v16bf negfrag(v16bf a) {
    FragU u;
    u.v = a;
#pragma unroll
    for (int j = 0; j < 16; ++j) u.s[j] ^= 0x8000u;
    return u.v;
}

// ---- CDNA5 async global->LDS copy (ASYNCcnt) -------------------------------
__device__ inline void async_tile(const bf16_t* src, long src_ld,
                                  bf16_t* dstTile, int tid) {
#pragma unroll
    for (int e = tid; e < 512; e += 256) {
        int row = e >> 3, wi = e & 7;
        unsigned loff = (unsigned)(unsigned long long)(dstTile + row * LDT + wi * 8);
        const bf16_t* g = src + (long)row * src_ld + wi * 8;
        asm volatile("global_load_async_to_lds_b128 %0, %1, off"
                     :: "v"(loff), "v"(g) : "memory");
    }
}

__device__ inline void wait_async0() {
    asm volatile("s_wait_asynccnt 0x0" ::: "memory");
}

// ---- CDNA5 Tensor Data Mover: one DMA op moves a strided 64x64 bf16 tile
// into LDS with hardware row padding (32 DWORDs data + 4 DWORDs pad = LDT=72).
// D# per CDNA5 ISA. Issued by one wave.
__device__ inline void tdm_load_tile(const bf16_t* gsrc, bf16_t* dstTile,
                                     unsigned stride_elems) {
    unsigned long long ga = (unsigned long long)gsrc;
    unsigned ldsoff = (unsigned)(unsigned long long)dstTile;
    v4u g0;
    g0[0] = 1u;                                   // count=1, user mode
    g0[1] = ldsoff;                               // lds_addr
    g0[2] = (unsigned)ga;                         // global_addr[31:0]
    g0[3] = (unsigned)(ga >> 32) | (2u << 30);    // global_addr[56:32] | type=2
    v8u g1;
    g1[0] = (1u << 16)    // data_size = 2 bytes
          | (1u << 20)    // pad_enable
          | (4u << 22)    // pad_interval: code 4 -> every 32 DWORDs (one 64-elem row)
          | (3u << 25);   // pad_amount:   code 3 -> 4 DWORDs (8 elems) -> LDT=72
    g1[1] = (64u << 16);            // tensor_dim0 = 64 (bits 79:48, low part)
    g1[2] = 0u;                     // tensor_dim0 hi | tensor_dim1 lo
    g1[3] = 1u | (64u << 16);       // tensor_dim1 = 65536 | tile_dim0 = 64
    g1[4] = 64u;                    // tile_dim1 = 64, tile_dim2 = 0
    g1[5] = stride_elems;           // tensor_dim0_stride (data_size units)
    g1[6] = 0u;
    g1[7] = 0u;
    asm volatile("tensor_load_to_lds %0, %1" :: "s"(g0), "s"(g1) : "memory");
}

__device__ inline void wait_tensor0() {
#if __has_builtin(__builtin_amdgcn_s_wait_tensorcnt)
    __builtin_amdgcn_s_wait_tensorcnt((short)0);
#else
    asm volatile("s_wait_tensorcnt 0x0" ::: "memory");
#endif
}

// ---------------------------------------------------------------------------
// Kernel 0: convert x to bf16, transpose-convert 8 weight matrices, biases.
// ---------------------------------------------------------------------------
struct PrepArgs {
    const float* x;
    const float* w[8];
    const float* b[6];
    bf16_t* xb;
    bf16_t* wt;
    float*  bias6;
};

__global__ void prep_kernel(PrepArgs a) {
    long idx = (long)blockIdx.x * blockDim.x + threadIdx.x;
    if (idx < (long)TD) {
        a.xb[idx] = (bf16_t)a.x[idx];
        return;
    }
    idx -= (long)TD;
    const long wtot = 8l * DMODEL * DMODEL;
    if (idx < wtot) {
        int m = (int)(idx / (DMODEL * DMODEL));
        int e = (int)(idx % (DMODEL * DMODEL));
        int n = e / DMODEL, k = e % DMODEL;
        a.wt[(long)m * DMODEL * DMODEL + (long)n * DMODEL + k] =
            (bf16_t)a.w[m][(long)k * DMODEL + n];
        return;
    }
    idx -= wtot;
    if (idx < 6l * DMODEL) {
        int m = (int)(idx / DMODEL), n = (int)(idx % DMODEL);
        a.bias6[m * DMODEL + n] = a.b[m][n];
    }
}

// ---------------------------------------------------------------------------
// Kernel 1: six projection GEMMs, 16x64 per wave; B fragments in distinct
// registers so loads overlap WMMAs (partial loadcnt waits).
// ---------------------------------------------------------------------------
__global__ void qkv_gemm_kernel(const bf16_t* __restrict__ xb,
                                const bf16_t* __restrict__ wt,
                                const float* __restrict__ bias6,
                                float* __restrict__ qkv) {
    int wave = threadIdx.x >> 5, lane = threadIdx.x & 31;
    int gid  = blockIdx.x * 8 + wave;       // 6*128*8 = 6144 jobs
    int out  = gid >> 10;
    int rem  = gid & 1023;
    int rowt = rem >> 3, cb = rem & 7;

    const bf16_t* A  = xb + (long)rowt * 16 * DMODEL;
    const bf16_t* BT = wt + (long)out * DMODEL * DMODEL + (long)cb * 64 * DMODEL;

    v8f acc0 = zero8(), acc1 = zero8(), acc2 = zero8(), acc3 = zero8();
#pragma unroll 2
    for (int k0 = 0; k0 < DMODEL; k0 += 32) {
        FragU af = load_fragu(A + k0, DMODEL, lane);
        FragU b0 = load_fragu(BT + 0l * 16 * DMODEL + k0, DMODEL, lane);
        FragU b1 = load_fragu(BT + 1l * 16 * DMODEL + k0, DMODEL, lane);
        FragU b2 = load_fragu(BT + 2l * 16 * DMODEL + k0, DMODEL, lane);
        FragU b3 = load_fragu(BT + 3l * 16 * DMODEL + k0, DMODEL, lane);
        acc0 = wmma_bf16(af.v, b0.v, acc0);
        acc1 = wmma_bf16(af.v, b1.v, acc1);
        acc2 = wmma_bf16(af.v, b2.v, acc2);
        acc3 = wmma_bf16(af.v, b3.v, acc3);
    }
    int n = lane & 15, half = (lane >> 4) & 1;
    float* O = qkv + (long)out * TD;
    v8f accs[4] = {acc0, acc1, acc2, acc3};
#pragma unroll
    for (int j = 0; j < 4; ++j) {
        int col = cb * 64 + j * 16 + n;
        float bv = bias6[out * DMODEL + col];
#pragma unroll
        for (int r = 0; r < 8; ++r) {
            int row = rowt * 16 + r + 8 * half;
            O[(long)row * DMODEL + col] = accs[j][r] + bv;
        }
    }
}

// ---------------------------------------------------------------------------
// Kernel 2: elementwise gates/phases + layout shuffles.
// ---------------------------------------------------------------------------
struct DeriveArgs {
    const float* qkv;
    bf16_t *qf_r, *qf_i, *kk_r, *kk_i, *qg, *kg;
    bf16_t *vT_r, *vT_i, *kkT_r, *kkT_i;
};

__global__ void derive_kernel(DeriveArgs a) {
    long idx = (long)blockIdx.x * blockDim.x + threadIdx.x;
    if (idx >= (long)TD) return;
    int t = (int)(idx / DMODEL), nc = (int)(idx % DMODEL);
    int h = nc >> 6, d = nc & 63;

    const float* Qr = a.qkv + 0l * TD;
    const float* Qi = a.qkv + 1l * TD;
    const float* Kr = a.qkv + 2l * TD;
    const float* Ki = a.qkv + 3l * TD;
    const float* Vr = a.qkv + 4l * TD;
    const float* Vi = a.qkv + 5l * TD;

    float qr = Qr[idx], qi = Qi[idx];
    float qgate = sqrtf(qr * qr + qi * qi);
    float qs = qgate / (qgate + EPSV);

    float kr = Kr[idx], ki = Ki[idx];
    float kgate = sqrtf(kr * kr + ki * ki);
    float ks = kgate / (kgate + EPSV);
    float kkr = kr * ks, kki = -ki * ks;

    long ih = ((long)h * T_SEQ + t) * HDIM + d;
    a.qf_r[ih] = (bf16_t)(qr * qs);
    a.qf_i[ih] = (bf16_t)(qi * qs);
    a.qg[ih]   = (bf16_t)qgate;
    a.kk_r[ih] = (bf16_t)kkr;
    a.kk_i[ih] = (bf16_t)kki;
    a.kg[ih]   = (bf16_t)kgate;

    long ihT = ((long)h * HDIM + d) * T_SEQ + t;
    a.kkT_r[ihT] = (bf16_t)kkr;
    a.kkT_i[ihT] = (bf16_t)kki;
    a.vT_r[ihT]  = (bf16_t)Vr[idx];
    a.vT_i[ihT]  = (bf16_t)Vi[idx];
}

// ---------------------------------------------------------------------------
// Kernel 3: chunked complex linear attention. Contiguous tiles are staged
// via per-lane async-to-LDS (ASYNCcnt); strided transposed tiles via the
// Tensor Data Mover (TENSORcnt), one descriptor per tile, wave 0 only.
// ---------------------------------------------------------------------------
struct AttnArgs {
    const bf16_t *qf_r, *qf_i, *kk_r, *kk_i, *qg, *kg;
    const bf16_t *vT_r, *vT_i, *kkT_r, *kkT_i;
    bf16_t *y_r, *y_i;
    const float *log_decay_s, *log_decay_z, *phase;
};

#define TQFR 0
#define TQFI 1
#define TQG  2
#define TKKR 3
#define TKKI 4
#define TKG  5
#define TVTR 6
#define TVTI 7
#define TKTR 8
#define TKTI 9

__global__ void __launch_bounds__(256) attn_kernel(AttnArgs a) {
    __shared__ __attribute__((aligned(16))) bf16_t stage[2][10][TILE_E];
    __shared__ __attribute__((aligned(16))) float  S_r[64 * LDT];
    __shared__ __attribute__((aligned(16))) float  S_i[64 * LDT];
    __shared__ __attribute__((aligned(16))) bf16_t A_r[64 * LDT];
    __shared__ __attribute__((aligned(16))) bf16_t A_i[64 * LDT];
    __shared__ __attribute__((aligned(16))) float  Gm[64 * LDT];
    __shared__ float Ztab[2][64];
    __shared__ float den_s[64];
    __shared__ float mr_tab[65], mi_tab[65], mz_tab[65];

    int h = blockIdx.x;
    int tid = threadIdx.x;
    int wave = tid >> 5, lane = tid & 31;
    int n = lane & 15, half = (lane >> 4) & 1;

    const long hbase  = (long)h * T_SEQ * HDIM;
    const long hTbase = (long)h * HDIM * T_SEQ;

    // prefetch chunk 0 while we initialize state
    {
        const bf16_t* rm[6] = {a.qf_r + hbase, a.qf_i + hbase, a.qg + hbase,
                               a.kk_r + hbase, a.kk_i + hbase, a.kg + hbase};
#pragma unroll
        for (int m = 0; m < 6; ++m) async_tile(rm[m], HDIM, &stage[0][m][0], tid);
        if (wave == 0) {
            tdm_load_tile(a.vT_r + hTbase,  &stage[0][TVTR][0], T_SEQ);
            tdm_load_tile(a.vT_i + hTbase,  &stage[0][TVTI][0], T_SEQ);
            tdm_load_tile(a.kkT_r + hTbase, &stage[0][TKTR][0], T_SEQ);
            tdm_load_tile(a.kkT_i + hTbase, &stage[0][TKTI][0], T_SEQ);
        }
    }

    float lr = -log1pf(expf(a.log_decay_s[h]));
    float lz = -log1pf(expf(a.log_decay_z[h]));
    float th = a.phase[h];

    for (int i = tid; i < 64 * LDT; i += 256) { S_r[i] = 0.f; S_i[i] = 0.f; }
    if (tid < 64) { Ztab[0][tid] = 0.f; Ztab[1][tid] = 0.f; }
    if (tid < 65) {
        float dd = (float)tid;
        float e = expf(dd * lr);
        mr_tab[tid] = e * cosf(dd * th);
        mi_tab[tid] = e * sinf(dd * th);
        mz_tab[tid] = expf(dd * lz);
    }
    if (wave == 0) wait_tensor0();
    wait_async0();
    __syncthreads();

    for (int c = 0; c < NCHUNK; ++c) {
        int t0 = c * 64;
        int cur = c & 1, nxt = cur ^ 1;
        const bf16_t (*st)[TILE_E] = stage[cur];

        // async + TDM prefetch of next chunk overlaps this chunk's compute
        if (c + 1 < NCHUNK) {
            long t1 = (long)(t0 + 64);
            const bf16_t* rm[6] = {a.qf_r + hbase + t1 * HDIM, a.qf_i + hbase + t1 * HDIM,
                                   a.qg + hbase + t1 * HDIM,   a.kk_r + hbase + t1 * HDIM,
                                   a.kk_i + hbase + t1 * HDIM, a.kg + hbase + t1 * HDIM};
#pragma unroll
            for (int m = 0; m < 6; ++m) async_tile(rm[m], HDIM, &stage[nxt][m][0], tid);
            if (wave == 0) {
                tdm_load_tile(a.vT_r + hTbase + t1,  &stage[nxt][TVTR][0], T_SEQ);
                tdm_load_tile(a.vT_i + hTbase + t1,  &stage[nxt][TVTI][0], T_SEQ);
                tdm_load_tile(a.kkT_r + hTbase + t1, &stage[nxt][TKTR][0], T_SEQ);
                tdm_load_tile(a.kkT_i + hTbase + t1, &stage[nxt][TKTI][0], T_SEQ);
            }
        }

        // ---- phase 1: scores (complex) + gate Gram, mask with alpha^(t-s)
        for (int sti = 0; sti < 2; ++sti) {
            int stt = wave + sti * 8;
            int rowt = stt >> 2, colt = stt & 3;
            v8f ar = zero8(), ai = zero8(), ag = zero8();
            if (rowt >= colt) {   // subtiles strictly above diagonal are all-zero
#pragma unroll
                for (int k0 = 0; k0 < 64; k0 += 32) {
                    int ab = rowt * 16 * LDT + k0;
                    int bb = colt * 16 * LDT + k0;
                    FragU aQr = load_fragu(&st[TQFR][ab], LDT, lane);
                    FragU aQi = load_fragu(&st[TQFI][ab], LDT, lane);
                    FragU aQg = load_fragu(&st[TQG][ab],  LDT, lane);
                    FragU bKr = load_fragu(&st[TKKR][bb], LDT, lane);
                    FragU bKi = load_fragu(&st[TKKI][bb], LDT, lane);
                    FragU bKg = load_fragu(&st[TKG][bb],  LDT, lane);
                    v16bf bKiN = negfrag(bKi.v);
                    ar = wmma_bf16(aQr.v, bKr.v, ar);
                    ar = wmma_bf16(aQi.v, bKiN,  ar);
                    ai = wmma_bf16(aQr.v, bKi.v, ai);
                    ai = wmma_bf16(aQi.v, bKr.v, ai);
                    ag = wmma_bf16(aQg.v, bKg.v, ag);
                }
            }
            int is = colt * 16 + n;
#pragma unroll
            for (int r = 0; r < 8; ++r) {
                int it = rowt * 16 + r + 8 * half;
                int dlt = it - is;
                float xr = ar[r], xi = ai[r], g = ag[r];
                if (dlt < 0) { xr = 0.f; xi = 0.f; g = 0.f; }
                else {
                    float wr = mr_tab[dlt], wi = mi_tab[dlt];
                    float tr = xr * wr - xi * wi;
                    float ti = xr * wi + xi * wr;
                    xr = tr; xi = ti;
                    g *= mz_tab[dlt];
                }
                A_r[it * LDT + is] = (bf16_t)xr;
                A_i[it * LDT + is] = (bf16_t)xi;
                Gm[it * LDT + is]  = g;
            }
        }
        __syncthreads();

        // ---- phase 1.5: den(t) = rowsum(G) + alpha_z^(i+1) Z.qg ; Z update
        if (tid < 64) {
            float s = 0.f;
            for (int j = 0; j < 64; ++j) s += Gm[tid * LDT + j];
            float zd = 0.f;
            const bf16_t* qrow = &st[TQG][tid * LDT];
            for (int k = 0; k < 64; ++k) zd += Ztab[cur][k] * (float)qrow[k];
            den_s[tid] = s + mz_tab[tid + 1] * zd;
        } else if (tid < 128) {
            int k = tid - 64;
            float acc = mz_tab[64] * Ztab[cur][k];
            for (int i = 0; i < 64; ++i)
                acc += mz_tab[63 - i] * (float)st[TKG][i * LDT + k];
            Ztab[nxt][k] = acc;
        }
        __syncthreads();

        // ---- phase 2: num = A@V (intra) + alpha^(i+1) * (qf @ S^T) (inter)
        for (int sti = 0; sti < 2; ++sti) {
            int stt = wave + sti * 8;
            int rowt = stt >> 2, colt = stt & 3;
            v8f nr = zero8(), ni = zero8(), pr = zero8(), pi = zero8();
#pragma unroll
            for (int k0 = 0; k0 < 64; k0 += 32) {
                FragU aAr = load_fragu(A_r + rowt * 16 * LDT + k0, LDT, lane);
                FragU aAi = load_fragu(A_i + rowt * 16 * LDT + k0, LDT, lane);
                int vb = colt * 16 * LDT + k0;
                FragU bVr = load_fragu(&st[TVTR][vb], LDT, lane);
                FragU bVi = load_fragu(&st[TVTI][vb], LDT, lane);
                v16bf bViN = negfrag(bVi.v);
                nr = wmma_bf16(aAr.v, bVr.v, nr);
                nr = wmma_bf16(aAi.v, bViN,  nr);
                ni = wmma_bf16(aAr.v, bVi.v, ni);
                ni = wmma_bf16(aAi.v, bVr.v, ni);

                int qb = rowt * 16 * LDT + k0;
                FragU aQr = load_fragu(&st[TQFR][qb], LDT, lane);
                FragU aQi = load_fragu(&st[TQFI][qb], LDT, lane);
                FragU bSr = load_fragu_f32(S_r + colt * 16 * LDT + k0, LDT, lane);
                FragU bSi = load_fragu_f32(S_i + colt * 16 * LDT + k0, LDT, lane);
                v16bf bSiN = negfrag(bSi.v);
                pr = wmma_bf16(aQr.v, bSr.v, pr);
                pr = wmma_bf16(aQi.v, bSiN,  pr);
                pi = wmma_bf16(aQr.v, bSi.v, pi);
                pi = wmma_bf16(aQi.v, bSr.v, pi);
            }
            int dv = colt * 16 + n;
#pragma unroll
            for (int r = 0; r < 8; ++r) {
                int it = rowt * 16 + r + 8 * half;
                float er = mr_tab[it + 1], ei = mi_tab[it + 1];
                float numr = nr[r] + er * pr[r] - ei * pi[r];
                float numi = ni[r] + er * pi[r] + ei * pr[r];
                float dinv = 1.f / (den_s[it] + EPSV);
                long yidx = (long)(t0 + it) * DMODEL + h * HDIM + dv;
                a.y_r[yidx] = (bf16_t)(numr * dinv);
                a.y_i[yidx] = (bf16_t)(numi * dinv);
            }
        }
        __syncthreads();

        // ---- phase 3: S = alpha^64 * S + sum_i alpha^(63-i) v_i kk_i^T
        for (int sti = 0; sti < 2; ++sti) {
            int stt = wave + sti * 8;
            int rowt = stt >> 2, colt = stt & 3;
            v8f ur = zero8(), ui = zero8();
#pragma unroll
            for (int k0 = 0; k0 < 64; k0 += 32) {
                int vb = rowt * 16 * LDT + k0;
                FragU fr = load_fragu(&st[TVTR][vb], LDT, lane);
                FragU fi = load_fragu(&st[TVTI][vb], LDT, lane);
                FragU var, vai, vaiN;
#pragma unroll
                for (int j = 0; j < 16; ++j) {
                    int il = k0 + (j >> 3) * 16 + half * 8 + (j & 7);
                    float wr = mr_tab[63 - il], wi = mi_tab[63 - il];
                    float vr = (float)fr.e[j], vi = (float)fi.e[j];
                    float xr = wr * vr - wi * vi;
                    float xi = wr * vi + wi * vr;
                    var.e[j]  = (bf16_t)xr;
                    vai.e[j]  = (bf16_t)xi;
                    vaiN.e[j] = (bf16_t)(-xi);
                }
                int kb = colt * 16 * LDT + k0;
                FragU bKr = load_fragu(&st[TKTR][kb], LDT, lane);
                FragU bKi = load_fragu(&st[TKTI][kb], LDT, lane);
                ur = wmma_bf16(var.v,  bKr.v, ur);
                ur = wmma_bf16(vaiN.v, bKi.v, ur);
                ui = wmma_bf16(var.v,  bKi.v, ui);
                ui = wmma_bf16(vai.v,  bKr.v, ui);
            }
            int kcol = colt * 16 + n;
            float cr = mr_tab[64], ci = mi_tab[64];
#pragma unroll
            for (int r = 0; r < 8; ++r) {
                int dv = rowt * 16 + r + 8 * half;
                int idx = dv * LDT + kcol;
                float sr0 = S_r[idx], si0 = S_i[idx];
                S_r[idx] = cr * sr0 - ci * si0 + ur[r];
                S_i[idx] = cr * si0 + ci * sr0 + ui[r];
            }
        }
        if (wave == 0) wait_tensor0();
        wait_async0();
        __syncthreads();
    }
}

// ---------------------------------------------------------------------------
// Kernel 4: complex output projection, 16x64 per wave, interleaved complex64.
// ---------------------------------------------------------------------------
__global__ void oproj_kernel(const bf16_t* __restrict__ y_r,
                             const bf16_t* __restrict__ y_i,
                             const bf16_t* __restrict__ wt_or,
                             const bf16_t* __restrict__ wt_oi,
                             const float* __restrict__ o_br,
                             const float* __restrict__ o_bi,
                             float* __restrict__ outp) {
    int wave = threadIdx.x >> 5, lane = threadIdx.x & 31;
    int gid  = blockIdx.x * 8 + wave;      // 128*8 = 1024 jobs
    int rowt = gid >> 3, cb = gid & 7;

    const bf16_t* Ar = y_r + (long)rowt * 16 * DMODEL;
    const bf16_t* Ai = y_i + (long)rowt * 16 * DMODEL;
    const bf16_t* Br = wt_or + (long)cb * 64 * DMODEL;
    const bf16_t* Bi = wt_oi + (long)cb * 64 * DMODEL;

    v8f accr[4] = {zero8(), zero8(), zero8(), zero8()};
    v8f acci[4] = {zero8(), zero8(), zero8(), zero8()};
#pragma unroll 2
    for (int k0 = 0; k0 < DMODEL; k0 += 32) {
        FragU ayr = load_fragu(Ar + k0, DMODEL, lane);
        FragU ayi = load_fragu(Ai + k0, DMODEL, lane);
        FragU br0 = load_fragu(Br + 0l * 16 * DMODEL + k0, DMODEL, lane);
        FragU bi0 = load_fragu(Bi + 0l * 16 * DMODEL + k0, DMODEL, lane);
        FragU br1 = load_fragu(Br + 1l * 16 * DMODEL + k0, DMODEL, lane);
        FragU bi1 = load_fragu(Bi + 1l * 16 * DMODEL + k0, DMODEL, lane);
        FragU br2 = load_fragu(Br + 2l * 16 * DMODEL + k0, DMODEL, lane);
        FragU bi2 = load_fragu(Bi + 2l * 16 * DMODEL + k0, DMODEL, lane);
        FragU br3 = load_fragu(Br + 3l * 16 * DMODEL + k0, DMODEL, lane);
        FragU bi3 = load_fragu(Bi + 3l * 16 * DMODEL + k0, DMODEL, lane);
        FragU* brj[4] = {&br0, &br1, &br2, &br3};
        FragU* bij[4] = {&bi0, &bi1, &bi2, &bi3};
#pragma unroll
        for (int j = 0; j < 4; ++j) {
            v16bf boiN = negfrag(bij[j]->v);
            accr[j] = wmma_bf16(ayr.v, brj[j]->v, accr[j]);
            accr[j] = wmma_bf16(ayi.v, boiN,      accr[j]);
            acci[j] = wmma_bf16(ayr.v, bij[j]->v, acci[j]);
            acci[j] = wmma_bf16(ayi.v, brj[j]->v, acci[j]);
        }
    }
    int n = lane & 15, half = (lane >> 4) & 1;
#pragma unroll
    for (int j = 0; j < 4; ++j) {
        int col = cb * 64 + j * 16 + n;
        float br = o_br[col], bi = o_bi[col];
#pragma unroll
        for (int r = 0; r < 8; ++r) {
            int row = rowt * 16 + r + 8 * half;
            long e = ((long)row * DMODEL + col) * 2;
            outp[e]     = accr[j][r] + br;
            outp[e + 1] = acci[j][r] + bi;
        }
    }
}

// ---------------------------------------------------------------------------
extern "C" void kernel_launch(void* const* d_in, const int* in_sizes, int n_in,
                              void* d_out, int out_size, void* d_ws, size_t ws_size,
                              hipStream_t stream) {
    char* ws = (char*)d_ws;
    size_t off = 0;
    auto take = [&](size_t bytes) -> char* {
        off = (off + 255) & ~(size_t)255;
        char* p = ws + off;
        off += bytes;
        return p;
    };

    bf16_t* xb    = (bf16_t*)take((size_t)TD * 2);
    bf16_t* wt    = (bf16_t*)take(8ull * DMODEL * DMODEL * 2);
    float*  bias6 = (float*)take(6ull * DMODEL * 4);
    float*  qkv   = (float*)take(6ull * TD * 4);
    bf16_t* qf_r  = (bf16_t*)take((size_t)TD * 2);
    bf16_t* qf_i  = (bf16_t*)take((size_t)TD * 2);
    bf16_t* kk_r  = (bf16_t*)take((size_t)TD * 2);
    bf16_t* kk_i  = (bf16_t*)take((size_t)TD * 2);
    bf16_t* qg    = (bf16_t*)take((size_t)TD * 2);
    bf16_t* kg    = (bf16_t*)take((size_t)TD * 2);
    bf16_t* vT_r  = (bf16_t*)take((size_t)TD * 2);
    bf16_t* vT_i  = (bf16_t*)take((size_t)TD * 2);
    bf16_t* kkT_r = (bf16_t*)take((size_t)TD * 2);
    bf16_t* kkT_i = (bf16_t*)take((size_t)TD * 2);
    bf16_t* y_r   = (bf16_t*)take((size_t)TD * 2);
    bf16_t* y_i   = (bf16_t*)take((size_t)TD * 2);

    PrepArgs pa;
    pa.x = (const float*)d_in[0];
    const int widx[8] = {1, 2, 5, 6, 9, 10, 13, 14};
    const int bidx[6] = {3, 4, 7, 8, 11, 12};
    for (int i = 0; i < 8; ++i) pa.w[i] = (const float*)d_in[widx[i]];
    for (int i = 0; i < 6; ++i) pa.b[i] = (const float*)d_in[bidx[i]];
    pa.xb = xb; pa.wt = wt; pa.bias6 = bias6;
    long prep_items = (long)TD + 8l * DMODEL * DMODEL + 6l * DMODEL;
    prep_kernel<<<(int)((prep_items + 255) / 256), 256, 0, stream>>>(pa);

    qkv_gemm_kernel<<<768, 256, 0, stream>>>(xb, wt, bias6, qkv);

    DeriveArgs da;
    da.qkv = qkv;
    da.qf_r = qf_r; da.qf_i = qf_i; da.kk_r = kk_r; da.kk_i = kk_i;
    da.qg = qg; da.kg = kg;
    da.vT_r = vT_r; da.vT_i = vT_i; da.kkT_r = kkT_r; da.kkT_i = kkT_i;
    derive_kernel<<<TD / 256, 256, 0, stream>>>(da);

    AttnArgs aa;
    aa.qf_r = qf_r; aa.qf_i = qf_i; aa.kk_r = kk_r; aa.kk_i = kk_i;
    aa.qg = qg; aa.kg = kg;
    aa.vT_r = vT_r; aa.vT_i = vT_i; aa.kkT_r = kkT_r; aa.kkT_i = kkT_i;
    aa.y_r = y_r; aa.y_i = y_i;
    aa.log_decay_s = (const float*)d_in[17];
    aa.log_decay_z = (const float*)d_in[18];
    aa.phase       = (const float*)d_in[19];
    attn_kernel<<<NH, 256, 0, stream>>>(aa);

    oproj_kernel<<<128, 256, 0, stream>>>(y_r, y_i, wt + 6ull * DMODEL * DMODEL,
                                          wt + 7ull * DMODEL * DMODEL,
                                          (const float*)d_in[15],
                                          (const float*)d_in[16],
                                          (float*)d_out);
}
